// GCNLayer_79706003079298
// MI455X (gfx1250) — matrix operand order repaired and checked by reading, not compiled, not run on hardware
//
#include <hip/hip_runtime.h>
#include <cstdint>

// GCN layer for MI455X (gfx1250, wave32, WMMA).
//   out = (A @ (X@W + b)) / deg,  A = (adj != 0), deg = colsum(A)
// N=8192, F=512. adj (256MB fp32) dominates HBM traffic -> stream it exactly
// once: each workgroup owns 64 output rows x ALL 512 columns. hidden is kept
// bf16 *transposed* (hiddenT[f][node], 8MB, L2-resident) so that WMMA
// B-fragments are contiguous 32B runs in LDS (2x ds_load_b128, no repacking).
// LDS tiles are double-buffered: next chunk staged via async-to-LDS + a
// register-held adj load while the current chunk's WMMAs execute.
// deg is fused into the aggregation pass via exact integer-valued f32 atomics.

#define N_NODES 8192
#define F_DIM   512
#define MT      64      // rows per workgroup in aggregation
#define KC      32      // K-chunk (one WMMA K) per LDS stage
#define HPADT   8       // pad per h_shT row: stride 40 u16 = 80B (16B-aligned)

typedef __bf16 bf16_t;
typedef __attribute__((ext_vector_type(16))) __bf16 v16bf;
typedef __attribute__((ext_vector_type(8)))  float  v8f;
typedef __attribute__((ext_vector_type(2)))  __bf16 v2bf;

// async-to-LDS builtin operand types (per hipcc diagnostic: int __vector(4))
typedef int v4i __attribute__((vector_size(16)));
typedef __attribute__((address_space(1))) v4i gv4i;   // global (__device__)
typedef __attribute__((address_space(3))) v4i lv4i;   // LDS (__shared__)

#if __has_builtin(__builtin_amdgcn_global_load_async_to_lds_b128) && \
    __has_builtin(__builtin_amdgcn_s_wait_asynccnt)
#define HAVE_ASYNC_LDS 1
#else
#define HAVE_ASYNC_LDS 0
#endif

__device__ __forceinline__ uint16_t bf16_rne_bits(float f) {
#if __has_builtin(__builtin_amdgcn_cvt_pk_bf16_f32)
    v2bf p = __builtin_amdgcn_cvt_pk_bf16_f32(f, 0.0f);
    return (uint16_t)(__builtin_bit_cast(uint32_t, p) & 0xFFFFu);
#else
    uint32_t u = __builtin_bit_cast(uint32_t, f);
    u += 0x7FFFu + ((u >> 16) & 1u);      // round-to-nearest-even
    return (uint16_t)(u >> 16);
#endif
}
__device__ __forceinline__ float bf16_bits_to_f32(uint16_t h) {
    return __builtin_bit_cast(float, (uint32_t)h << 16);
}
__device__ __forceinline__ bf16_t bits_to_bf16(uint16_t h) {
    return __builtin_bit_cast(bf16_t, h);
}

struct U16x16 { uint4 lo, hi; };
__device__ __forceinline__ v16bf make_v16bf(uint4 lo, uint4 hi) {
    U16x16 u{lo, hi};
    return __builtin_bit_cast(v16bf, u);   // 32B -> 8 VGPRs, no repack
}

// ---------------------------------------------------------------------------
// Kernel 0: zero the degree accumulator (ws is not re-poisoned between calls).
// ---------------------------------------------------------------------------
__global__ void gcn_zero_deg(float* __restrict__ deg) {
    int i = blockIdx.x * blockDim.x + threadIdx.x;
    if (i < N_NODES) deg[i] = 0.0f;
}

// ---------------------------------------------------------------------------
// Kernel 1: hiddenT[f][node] = (X @ W + b)^T, stored bf16 (RNE) in workspace.
// One wave per 16x16 output tile; split-bf16 (hi+lo) operands give ~fp32
// accuracy via 3 bf16 WMMAs per K=32 step (lo*lo term dropped).
// ---------------------------------------------------------------------------
__global__ void __launch_bounds__(256)
gcn_hidden_kernel(const float* __restrict__ X, const float* __restrict__ W,
                  const float* __restrict__ bias, uint16_t* __restrict__ hiddenT)
{
    const int lane = threadIdx.x & 31;
    const int wave = threadIdx.x >> 5;
    const int m0 = blockIdx.y * 16;
    const int n0 = (blockIdx.x * 8 + wave) * 16;

    const int lm   = lane & 15;
    const int kh   = lane >> 4;       // which K-half this lane holds
    const int mrow = m0 + lm;
    const int ncol = n0 + lm;

    v8f acc = {};
    for (int kk = 0; kk < F_DIM; kk += 32) {
        // A-operand (X tile, 16x32): lane holds row lm; elem e -> K=16*kh+e
        // (same K order as the B operand)
        v16bf xhi, xlo;
#pragma unroll
        for (int e = 0; e < 16; ++e) {
            int k = kk + kh * 16 + e;
            float x = X[(size_t)mrow * F_DIM + k];
            uint16_t hb = bf16_rne_bits(x);
            uint16_t lb = bf16_rne_bits(x - bf16_bits_to_f32(hb));
            xhi[e] = bits_to_bf16(hb);
            xlo[e] = bits_to_bf16(lb);
        }
        // B-operand (W tile, 32x16): lanes 0-15 K=0..15, lanes 16-31 K=16..31,
        // elem e -> K = 16*kh + e; column = lm
        v16bf whi, wlo;
        const int kb = kk + kh * 16;
#pragma unroll
        for (int e = 0; e < 16; ++e) {
            float w = W[(size_t)(kb + e) * F_DIM + ncol];
            uint16_t hb = bf16_rne_bits(w);
            uint16_t lb = bf16_rne_bits(w - bf16_bits_to_f32(hb));
            whi[e] = bits_to_bf16(hb);
            wlo[e] = bits_to_bf16(lb);
        }
        acc = __builtin_amdgcn_wmma_f32_16x16x32_bf16(false, xhi, false, whi,
                                                      (short)0, acc, false, false);
        acc = __builtin_amdgcn_wmma_f32_16x16x32_bf16(false, xlo, false, whi,
                                                      (short)0, acc, false, false);
        acc = __builtin_amdgcn_wmma_f32_16x16x32_bf16(false, xhi, false, wlo,
                                                      (short)0, acc, false, false);
    }
    // C/D layout: VGPR v holds M = v + 8*kh, N = lm. The 8 M values of one
    // lane are consecutive -> one 16B store into the transposed layout.
    const float bv = bias[ncol];
    uint32_t p0 = (uint32_t)bf16_rne_bits(acc[0] + bv) |
                  ((uint32_t)bf16_rne_bits(acc[1] + bv) << 16);
    uint32_t p1 = (uint32_t)bf16_rne_bits(acc[2] + bv) |
                  ((uint32_t)bf16_rne_bits(acc[3] + bv) << 16);
    uint32_t p2 = (uint32_t)bf16_rne_bits(acc[4] + bv) |
                  ((uint32_t)bf16_rne_bits(acc[5] + bv) << 16);
    uint32_t p3 = (uint32_t)bf16_rne_bits(acc[6] + bv) |
                  ((uint32_t)bf16_rne_bits(acc[7] + bv) << 16);
    uint4 pk = make_uint4(p0, p1, p2, p3);
    *(uint4*)(hiddenT + (size_t)ncol * N_NODES + m0 + 8 * kh) = pk;
}

// ---------------------------------------------------------------------------
// Kernel 2: out_unnorm = A @ hidden  (A binarized on the fly -> exact bf16 0/1)
// plus fused deg accumulation. 512 threads = 16 waves; block tile 64 x 512 so
// adj is streamed from HBM exactly once. Wave (wm,wn) owns 32 rows x 64 cols
// = 2x4 WMMA tiles. Fragments are pure ds_load_b128 pairs. Double-buffered
// staging: next hidden chunk via GLOBAL_LOAD_ASYNC_TO_LDS_B128 + next adj
// float4 into registers, overlapped with the current chunk's WMMAs.
// ---------------------------------------------------------------------------
__global__ void __launch_bounds__(512)
gcn_aggregate_kernel(const float* __restrict__ adj,
                     const uint16_t* __restrict__ hiddenT,
                     float* __restrict__ deg, float* __restrict__ out)
{
    __shared__ __align__(16) uint16_t a_sh[2][MT][KC];             // 8 KB
    __shared__ __align__(16) uint16_t h_shT[2][F_DIM][KC + HPADT]; // 80 KB

    const int tid  = threadIdx.x;
    const int lane = tid & 31;
    const int wave = tid >> 5;
    const int wm   = wave >> 3;     // 0..1  (32-row slab)
    const int wn   = wave & 7;      // 0..7  (64-col slab)
    const int m0   = blockIdx.x * MT;

    const int lm = lane & 15;
    const int kh = lane >> 4;

    v8f acc[2][4];
#pragma unroll
    for (int mt = 0; mt < 2; ++mt)
#pragma unroll
        for (int nt = 0; nt < 4; ++nt) { v8f z = {}; acc[mt][nt] = z; }

    // staging coordinates (per thread)
    const int ar = tid >> 3;            // 0..63 : adj row within tile
    const int ac = (tid & 7) * 4;       // 0..28 : adj col (float4)
    const float* adj_src = adj + (size_t)(m0 + ar) * N_NODES + ac;

    auto stage_h = [&](int kk, int buf) {
#pragma unroll
        for (int i = 0; i < 4; ++i) {
            int id = tid + i * 512;             // 0..2047 16B tiles
            int n  = id >> 2;                   // 0..511
            int q  = (id & 3) * 8;              // u16 offset within chunk
            const uint16_t* src = hiddenT + (size_t)n * N_NODES + kk + q;
            uint16_t* dst = &h_shT[buf][n][q];
#if HAVE_ASYNC_LDS
            __builtin_amdgcn_global_load_async_to_lds_b128(
                (gv4i*)const_cast<uint16_t*>(src), (lv4i*)dst, 0, 0);
#else
            *(uint4*)dst = *(const uint4*)src;
#endif
        }
    };
    auto pack_a = [&](const float4& v, int buf) {
        uint64_t pk = (uint64_t)((v.x != 0.0f) ? 0x3F80u : 0u)
                    | ((uint64_t)((v.y != 0.0f) ? 0x3F80u : 0u) << 16)
                    | ((uint64_t)((v.z != 0.0f) ? 0x3F80u : 0u) << 32)
                    | ((uint64_t)((v.w != 0.0f) ? 0x3F80u : 0u) << 48);
        *(uint64_t*)(&a_sh[buf][ar][ac]) = pk;
    };

    // ---- prologue: stage chunk 0 into buffer 0 ----
    stage_h(0, 0);
    pack_a(*(const float4*)(adj_src), 0);
#if HAVE_ASYNC_LDS
    __builtin_amdgcn_s_wait_asynccnt(0);
#endif
    __syncthreads();

    for (int kk = 0; kk < N_NODES; kk += KC) {
        const int cur = (kk >> 5) & 1;
        const int nxt = cur ^ 1;
        const bool has_next = (kk + KC) < N_NODES;

        // ---- issue next chunk's staging while we compute on `cur` ----
        float4 av;
        if (has_next) {
            stage_h(kk + KC, nxt);                       // async L2 -> LDS
            av = *(const float4*)(adj_src + kk + KC);    // adj -> registers
            if (kk + 2 * KC < N_NODES)
                __builtin_prefetch(adj_src + kk + 2 * KC, 0, 3);
        }

        // ---- fused degree: column sums of cur 64x32 slab (exact ints) ----
        if (tid < KC) {
            int s = 0;
#pragma unroll 8
            for (int r = 0; r < MT; ++r) s += (a_sh[cur][r][tid] != 0);
            atomicAdd(&deg[kk + tid], (float)s);
        }

        // ---- fragments (ds_load_b128 pairs) + WMMA on `cur` ----
        v16bf afr[2];
#pragma unroll
        for (int mt = 0; mt < 2; ++mt) {
            const uint16_t* ap = &a_sh[cur][wm * 32 + mt * 16 + lm][0];
            afr[mt] = make_v16bf(*(const uint4*)(ap + kh * 16),
                                 *(const uint4*)(ap + kh * 16 + 8));
        }
#pragma unroll
        for (int nt = 0; nt < 4; ++nt) {
            const int n = wn * 64 + nt * 16 + lm;
            const uint16_t* hp = &h_shT[cur][n][kh * 16];
            // B layout: lanes 0-15 K=0..15, lanes 16-31 K=16..31; elem e -> K+e
            const v16bf bfr = make_v16bf(*(const uint4*)(hp),
                                         *(const uint4*)(hp + 8));
#pragma unroll
            for (int mt = 0; mt < 2; ++mt)
                acc[mt][nt] = __builtin_amdgcn_wmma_f32_16x16x32_bf16(
                    false, afr[mt], false, bfr, (short)0, acc[mt][nt], false, false);
        }

        // ---- commit next A tile (safe: peers consumed `nxt` before the
        //      previous barrier; adj load latency hidden behind the WMMAs) ----
        if (has_next) pack_a(av, nxt);

#if HAVE_ASYNC_LDS
        __builtin_amdgcn_s_wait_asynccnt(0);   // next h chunk fully in LDS
#endif
        __syncthreads();
    }

    // ---- store unnormalized output (normalized by kernel 3) ----
#pragma unroll
    for (int mt = 0; mt < 2; ++mt)
#pragma unroll
        for (int nt = 0; nt < 4; ++nt) {
            const int n = wn * 64 + nt * 16 + lm;
#pragma unroll
            for (int v = 0; v < 8; ++v) {
                int m = m0 + wm * 32 + mt * 16 + v + 8 * kh;
                out[(size_t)m * F_DIM + n] = acc[mt][nt][v];
            }
        }
}

// ---------------------------------------------------------------------------
// Kernel 3: out[i, :] /= deg[i]
// ---------------------------------------------------------------------------
__global__ void gcn_normalize_kernel(float* __restrict__ out,
                                     const float* __restrict__ deg)
{
    int idx = blockIdx.x * blockDim.x + threadIdx.x;
    if (idx < N_NODES * F_DIM) {
        out[idx] = out[idx] / deg[idx >> 9];    // F_DIM == 512
    }
}

// ---------------------------------------------------------------------------
extern "C" void kernel_launch(void* const* d_in, const int* in_sizes, int n_in,
                              void* d_out, int out_size, void* d_ws, size_t ws_size,
                              hipStream_t stream)
{
    const float* X   = (const float*)d_in[0];   // [8192, 512]
    const float* adj = (const float*)d_in[1];   // [8192, 8192]
    const float* W   = (const float*)d_in[2];   // [512, 512]
    const float* b   = (const float*)d_in[3];   // [512]
    float* out = (float*)d_out;                 // [8192, 512]

    uint16_t* hiddenT = (uint16_t*)d_ws;        // [512][8192] bf16 = 8 MB
    float*    deg     = (float*)((char*)d_ws + (size_t)N_NODES * F_DIM * 2);

    gcn_zero_deg<<<N_NODES / 256, 256, 0, stream>>>(deg);

    dim3 gA(F_DIM / 16 / 8, N_NODES / 16);      // (4, 512), 8 waves/block
    gcn_hidden_kernel<<<gA, 256, 0, stream>>>(X, W, b, hiddenT);

    gcn_aggregate_kernel<<<N_NODES / MT, 512, 0, stream>>>(adj, hiddenT, deg, out);

    gcn_normalize_kernel<<<(N_NODES * F_DIM) / 256, 256, 0, stream>>>(out, deg);
}
